// CTE_37512244364030
// MI455X (gfx1250) — matrix operand order repaired and checked by reading, not compiled, not run on hardware
//
#include <hip/hip_runtime.h>
#include <hip/hip_bf16.h>

typedef __attribute__((ext_vector_type(16))) _Float16 v16h;
typedef __attribute__((ext_vector_type(8)))  float    v8f;

// GCC-style vector-of-4-int, matching the async-LDS builtin's pointee type
typedef int v4i_vs __attribute__((vector_size(16)));
typedef v4i_vs __attribute__((address_space(1)))* async_gptr;
typedef v4i_vs __attribute__((address_space(3)))* async_lptr;

#if __has_builtin(__builtin_amdgcn_global_load_async_to_lds_b128)
#define FERN_HAVE_ASYNC_LDS 1
#endif

#define FERN_BLOCK 256
#define ROWP 36   // LDS table row stride in floats (16B-aligned float4 chunks, bank-spread)

// ---------------------------------------------------------------------------
// Fern layer: gathers + sigmoid bits + top-2 flip expansion + LDS table votes.
// One thread per (b,h,w) pixel; loops over the 8 ferns, staging each fern's
// 256x32 float table slice into LDS (async DMA path when available).
// Produces votes (B,32,H,W).
// ---------------------------------------------------------------------------
__global__ __launch_bounds__(FERN_BLOCK) void fern_kernel(
    const float* __restrict__ X, const int* __restrict__ off,
    const int* __restrict__ ch, const float* __restrict__ th,
    const float* __restrict__ tab, float* __restrict__ votes,
    int B, int C, int H, int W)
{
    __shared__ float tabs[256 * ROWP];   // 36 KB: one fern's table, padded rows
    __shared__ float th_s[64];
    __shared__ int   off_s[256];
    __shared__ int   ch_s[128];

    const int tid = threadIdx.x;
    if (tid < 64)  th_s[tid] = th[tid];
    if (tid < 128) ch_s[tid] = ch[tid];
    off_s[tid] = off[tid];
    __syncthreads();

    const int HW = H * W;
    const int N  = B * HW;
    const int p  = blockIdx.x * FERN_BLOCK + tid;
    const bool valid = p < N;
    int b = 0, h = 0, w = 0;
    if (valid) { b = p / HW; int r = p - b * HW; h = r / W; w = r - h * W; }

    float acc[32];
#pragma unroll
    for (int d = 0; d < 32; ++d) acc[d] = 0.f;

    for (int m = 0; m < 8; ++m) {
        if (m < 7) __builtin_prefetch(tab + (m + 1) * 8192, 0, 0);

        // ---- stage tab[m] (256 rows x 32 floats) into LDS, rows padded ----
        {
            const float* g = tab + m * 8192;
#pragma unroll
            for (int i = 0; i < 8; ++i) {
                int c4  = tid + i * FERN_BLOCK;      // float4 chunk id, 0..2047
                int row = c4 >> 3;
                int col = (c4 & 7) << 2;
                const float* gsrc = g + row * 32 + col;
                float*       ldst = &tabs[row * ROWP + col];
#ifdef FERN_HAVE_ASYNC_LDS
                __builtin_amdgcn_global_load_async_to_lds_b128(
                    (async_gptr)gsrc, (async_lptr)ldst, 0, 0);
#else
                *(float4*)ldst = *(const float4*)gsrc;
#endif
            }
#ifdef FERN_HAVE_ASYNC_LDS
#if __has_builtin(__builtin_amdgcn_s_wait_asynccnt)
            __builtin_amdgcn_s_wait_asynccnt(0);
#else
            asm volatile("s_wait_asynccnt 0" ::: "memory");
#endif
#endif
        }
        __syncthreads();

        if (valid) {
            int   word = 0;
            float bp   = 1.f;
            float a0 = -1e30f, a1 = -1e30f;     // top-2 ambiguity values
            float q0v = 1.f, q1v = 1.f;         // pmax at those bits
            int   m0b = 0, m1b = 0;             // bit masks of those bits
#pragma unroll
            for (int k = 0; k < 8; ++k) {
                const int e = m * 8 + k;
                int dy0 = off_s[e * 4 + 0], dx0 = off_s[e * 4 + 1];
                int dy1 = off_s[e * 4 + 2], dx1 = off_s[e * 4 + 3];
                int c0 = ch_s[e * 2 + 0], c1 = ch_s[e * 2 + 1];
                int y0 = h + dy0, x0 = w + dx0;
                int y1 = h + dy1, x1 = w + dx1;
                float g0 = (y0 >= 0 && y0 < H && x0 >= 0 && x0 < W)
                               ? X[((b * C + c0) * H + y0) * W + x0] : 0.f;
                float g1 = (y1 >= 0 && y1 < H && x1 >= 0 && x1 < W)
                               ? X[((b * C + c1) * H + y1) * W + x1] : 0.f;
                float z = (g0 - g1 - th_s[e]) * 10.f;   // /TEMP
                float s = 1.f / (1.f + __expf(-z));
                bool  hard = s > 0.5f;
                float q = hard ? s : 1.f - s;
                bp *= q;
                if (hard) word |= (1 << k);
                float amb = -fabsf(s - 0.5f);
                if (amb > a0) { a1 = a0; q1v = q0v; m1b = m0b;
                                a0 = amb; q0v = q;  m0b = 1 << k; }
                else if (amb > a1) { a1 = amb; q1v = q; m1b = 1 << k; }
            }
            float r0 = (1.f - q0v) / fmaxf(q0v, 1e-6f);
            float r1 = (1.f - q1v) / fmaxf(q1v, 1e-6f);
            float pr[4];
            pr[0] = bp; pr[1] = bp * r0; pr[2] = bp * r1; pr[3] = bp * r0 * r1;
            float inv = 1.f / (pr[0] + pr[1] + pr[2] + pr[3]);
#pragma unroll
            for (int a = 0; a < 4; ++a) pr[a] *= inv;
            const float* t0 = &tabs[(word)             * ROWP];
            const float* t1 = &tabs[(word ^ m0b)       * ROWP];
            const float* t2 = &tabs[(word ^ m1b)       * ROWP];
            const float* t3 = &tabs[(word ^ (m0b|m1b)) * ROWP];
#pragma unroll
            for (int d = 0; d < 32; d += 4) {
                float4 u0 = *(const float4*)(t0 + d);
                float4 u1 = *(const float4*)(t1 + d);
                float4 u2 = *(const float4*)(t2 + d);
                float4 u3 = *(const float4*)(t3 + d);
                acc[d + 0] += pr[0]*u0.x + pr[1]*u1.x + pr[2]*u2.x + pr[3]*u3.x;
                acc[d + 1] += pr[0]*u0.y + pr[1]*u1.y + pr[2]*u2.y + pr[3]*u3.y;
                acc[d + 2] += pr[0]*u0.z + pr[1]*u1.z + pr[2]*u2.z + pr[3]*u3.z;
                acc[d + 3] += pr[0]*u0.w + pr[1]*u1.w + pr[2]*u2.w + pr[3]*u3.w;
            }
        }
        __syncthreads();   // table reads done before next fern overwrites LDS
    }

    if (valid) {
#pragma unroll
        for (int d = 0; d < 32; ++d)
            votes[((b * 32 + d) * H + h) * W + w] = acc[d];
    }
}

// ---------------------------------------------------------------------------
// 3x3 mean pool (VALID) on the WMMA pipe.
// Horizontal 3-sum on VALU into LDS tile T[18][16], then vertical band-matrix
// product Out[16x16] = S[16x18] * T[18x16] as one v_wmma_f32_16x16x32_f16.
// One wave per 16x16 output tile; 4 waves per block.
// ---------------------------------------------------------------------------
__global__ __launch_bounds__(128) void pool3_wmma_kernel(
    const float* __restrict__ V, float* __restrict__ O,
    int H, int W, int Ho, int Wo, int tilesX, int tilesY, int nWaves)
{
    __shared__ float T[4][18 * 16];
    const int tid  = threadIdx.x;
    const int lane = tid & 31;
    const int wsl  = tid >> 5;
    const int gw   = blockIdx.x * 4 + wsl;
    const int tiles = tilesX * tilesY;
    const bool wvalid = gw < nWaves;       // wave-uniform
    int b = 0, d = 0, oy = 0, ox = 0;
    if (wvalid) {
        int t  = gw % tiles;
        int bd = gw / tiles;
        b = bd >> 5; d = bd & 31;
        int ty = t / tilesX, tx = t - ty * tilesX;
        oy = ty * 16; ox = tx * 16;
    }
    const float* base = V + ((b * 32 + d) * H) * W;

    // horizontal 3-sum: T[r][n] = sum_j v[oy+r][ox+n+j], 18x16 entries
#pragma unroll
    for (int i = 0; i < 9; ++i) {
        int idx = lane + i * 32;          // 0..287
        int r = idx >> 4, n = idx & 15;
        int y = oy + r;
        float s = 0.f;
        if (wvalid && y < H) {
#pragma unroll
            for (int j = 0; j < 3; ++j) {
                int x = ox + n + j;
                if (x < W) s += base[y * W + x];
            }
        }
        T[wsl][idx] = s;
    }
    __syncthreads();

    // Build fragments per CDNA5 16-bit layouts (wave32).
    // A (16x32): lane<16: e<8 -> K=e, e>=8 -> K=e+8 ; lane>=16: +8 / +16.
    // B (32x16): N = lane%16, K same as A.
    const int Mr = lane & 15;
    const int hi = lane >> 4;
    v16h a{}, bf{};
#pragma unroll
    for (int e = 0; e < 16; ++e) {
        int Ka = (hi == 0) ? (e < 8 ? e : e + 8) : (e < 8 ? e + 8 : e + 16);
        a[e] = (_Float16)((Ka >= Mr && Ka <= Mr + 2) ? 1.0f : 0.0f);
        int Kb = Ka;
        bf[e] = (_Float16)((Kb < 18) ? T[wsl][Kb * 16 + Mr] : 0.0f);
    }
    v8f c = {};
    c = __builtin_amdgcn_wmma_f32_16x16x32_f16(
            /*neg_a=*/false, a, /*neg_b=*/false, bf,
            /*c_mod=*/(short)0, c, /*reuse_a=*/false, /*reuse_b=*/false);

#pragma unroll
    for (int r8 = 0; r8 < 8; ++r8) {
        int M = r8 + 8 * hi;              // C/D layout: VGPR r -> M = r + 8*(lane/16)
        int y = oy + M, x = ox + Mr;
        if (wvalid && y < Ho && x < Wo)
            O[((b * 32 + d) * Ho + y) * Wo + x] = c[r8] * (1.0f / 9.0f);
    }
}

// ---------------------------------------------------------------------------
extern "C" void kernel_launch(void* const* d_in, const int* in_sizes, int n_in,
                              void* d_out, int out_size, void* d_ws, size_t ws_size,
                              hipStream_t stream)
{
    (void)in_sizes; (void)n_in; (void)out_size; (void)ws_size;

    const float* x = (const float*)d_in[0];
    float* wsf   = (float*)d_ws;
    float* votes = wsf;                          // max 8*32*96*96 floats
    float* feat1 = votes + 8 * 32 * 96 * 96;     // (8,32,94,94)
    float* feat2 = feat1 + 8 * 32 * 94 * 94;     // (8,32,92,92)
    float* feat3 = feat2 + 8 * 32 * 92 * 92;     // (8,32,90,90)

    const int B = 8;
    const int Hs[4] = {96, 94, 92, 90};
    const int Cs[4] = {8, 32, 32, 32};
    const float* ins[4]  = {x, feat1, feat2, feat3};
    float*       outs[4] = {feat1, feat2, feat3, (float*)d_out};

    for (int l = 0; l < 4; ++l) {
        const int H = Hs[l], W = Hs[l], C = Cs[l];
        const int*   off = (const int*)  d_in[1 + 4 * l];
        const int*   chp = (const int*)  d_in[2 + 4 * l];
        const float* th  = (const float*)d_in[3 + 4 * l];
        const float* tab = (const float*)d_in[4 + 4 * l];

        int N = B * H * W;
        fern_kernel<<<(N + FERN_BLOCK - 1) / FERN_BLOCK, FERN_BLOCK, 0, stream>>>(
            ins[l], off, chp, th, tab, votes, B, C, H, W);

        int Ho = H - 2, Wo = W - 2;
        int tX = (Wo + 15) / 16, tY = (Ho + 15) / 16;
        int nW = B * 32 * tX * tY;
        pool3_wmma_kernel<<<(nW + 3) / 4, 128, 0, stream>>>(
            votes, outs[l], H, W, Ho, Wo, tX, tY, nW);
    }
}
